// CausalMultiHeadSelfAttention_369367188058
// MI455X (gfx1250) — matrix operand order repaired and checked by reading, not compiled, not run on hardware
//
#include <hip/hip_runtime.h>
#include <hip/hip_bf16.h>

#define B_  2
#define T_  2048
#define D_  1024
#define H_  16
#define HD_ 64

typedef __attribute__((ext_vector_type(16))) __bf16 v16bf;
typedef __attribute__((ext_vector_type(8)))  float  v8f;
typedef __attribute__((ext_vector_type(4)))  unsigned int u32x4;
typedef __attribute__((ext_vector_type(8)))  int i32x8;
typedef __attribute__((ext_vector_type(4)))  int i32x4;

union F16x16 { v16bf v; uint4 u[2]; };

// Tensor Data Mover: gate purely on builtin availability; pick arity by clang major
// (ROCm clang-22: 5 args; upstream clang-23+: 6 args).
#if __has_builtin(__builtin_amdgcn_tensor_load_to_lds)
#define USE_TDM 1
#else
#define USE_TDM 0
#endif

static __device__ __forceinline__ v8f wmma_bf16(v16bf a, v16bf b, v8f c) {
  return __builtin_amdgcn_wmma_f32_16x16x32_bf16(false, a, false, b, (short)0, c, false, false);
}
static __device__ __forceinline__ v8f zero8() {
  v8f z = {0.f,0.f,0.f,0.f,0.f,0.f,0.f,0.f};
  return z;
}

#if USE_TDM
// Issue one TDM DMA: 32(K) x 128(rows) bf16 tile, row stride K elements,
// global -> LDS (contiguous 128*32*2 bytes at ldsOff).
static __device__ __forceinline__ void tdm_load_tile(const __bf16* gsrc,
                                                     unsigned ldsOff, int K) {
  unsigned long long ga = (unsigned long long)(__UINTPTR_TYPE__)gsrc;
  u32x4 g0;
  g0[0] = 1u;                                        // count=1 (valid D#)
  g0[1] = ldsOff;                                    // lds_addr (bytes)
  g0[2] = (unsigned)(ga & 0xFFFFFFFFu);              // global_addr[31:0]
  g0[3] = (unsigned)((ga >> 32) & 0x1FFFFFFu)        // global_addr[56:32]
        | (2u << 30);                                // type = 2 ("image")
  const unsigned td0 = (unsigned)K;                  // tensor_dim0
  const unsigned td1 = 1u << 20;                     // tensor_dim1 (no OOB)
  const unsigned long long st0 = (unsigned long long)K; // dim0 stride
  i32x8 g1;
  g1[0] = (int)0x00010000u;                          // data_size=1 (2 bytes)
  g1[1] = (int)(td0 << 16);                          // tensor_dim0[15:0] @bit48
  g1[2] = (int)((td0 >> 16) | (td1 << 16));          // td0 hi | td1 lo
  g1[3] = (int)((td1 >> 16) | (32u << 16));          // td1 hi | tile_dim0=32
  g1[4] = (int)128u;                                 // tile_dim1=128, tile_dim2=0
  g1[5] = (int)(unsigned)(st0 & 0xFFFFFFFFu);        // td0_stride[31:0]
  g1[6] = (int)(unsigned)((st0 >> 32) & 0xFFFFu);    // td0_stride hi | td1_stride lo
  g1[7] = 0;                                         // td1_stride hi
  i32x4 gz = {0, 0, 0, 0};
#if defined(__clang_major__) && (__clang_major__ >= 23)
  i32x8 z8 = {0, 0, 0, 0, 0, 0, 0, 0};
  __builtin_amdgcn_tensor_load_to_lds(g0, g1, gz, gz, z8, 0);
#else
  __builtin_amdgcn_tensor_load_to_lds(g0, g1, gz, gz, 0);
#endif
}
#endif

// ---------------- conversion kernels ----------------

__global__ void cvt_f32_bf16(const float* __restrict__ in, __bf16* __restrict__ out, int n) {
  int i = blockIdx.x * blockDim.x + threadIdx.x;
  if (i < n) out[i] = (__bf16)in[i];
}

// out[n*D + k] = in[k*D + n]  (bf16 transposed weight, [N][K] row-major)
__global__ void cvt_w_transpose(const float* __restrict__ in, __bf16* __restrict__ out) {
  int i = blockIdx.x * blockDim.x + threadIdx.x; // linear over out
  int n = i >> 10;          // D_=1024
  int k = i & (D_ - 1);
  out[i] = (__bf16)in[k * D_ + n];
}

// ---------------- 128x128 WMMA GEMM  C = A[M,K] * B[K,N], B given transposed [N,K] ----------------
// MODE 0: Q  -> bf16 [B,H,T,HD], scaled by 1/sqrt(HD)
// MODE 1: K  -> bf16 [B,H,T,HD]
// MODE 2: V  -> bf16 [B,H,HD,T] (transposed per head)
// MODE 3: O  -> fp32 [M,N]
template <int MODE>
__global__ __launch_bounds__(256) void gemm_bf16_128x128(
    const __bf16* __restrict__ A,
    const __bf16* __restrict__ Bt,
    void* __restrict__ Out,
    int M, int N, int K)
{
#if USE_TDM
  __shared__ __bf16 As[2][128][32];
  __shared__ __bf16 Bs[2][128][32];
#else
  __shared__ __bf16 As[1][128][32];
  __shared__ __bf16 Bs[1][128][32];
#endif

  const int tid  = threadIdx.x;
  const int lane = tid & 31, wave = tid >> 5;
  const int half = lane >> 4, l15 = lane & 15;
  const int wm = wave & 1;   // 2 row groups of 64
  const int wn = wave >> 1;  // 4 col groups of 32
  const int m0 = blockIdx.y * 128, n0 = blockIdx.x * 128;

  v8f acc[4][2];
#pragma unroll
  for (int i = 0; i < 4; ++i)
#pragma unroll
    for (int j = 0; j < 2; ++j) acc[i][j] = zero8();

#if USE_TDM
  // ---- double-buffered TDM pipeline: wave 0 drives the Tensor Data Mover ----
  const bool issuer = (wave == 0);
  unsigned offA[2], offB[2];
  offA[0] = (unsigned)(__UINTPTR_TYPE__)&As[0][0][0];
  offA[1] = (unsigned)(__UINTPTR_TYPE__)&As[1][0][0];
  offB[0] = (unsigned)(__UINTPTR_TYPE__)&Bs[0][0][0];
  offB[1] = (unsigned)(__UINTPTR_TYPE__)&Bs[1][0][0];
  const int nk = K >> 5;
  if (issuer) {
    tdm_load_tile(A  + (size_t)m0 * K, offA[0], K);
    tdm_load_tile(Bt + (size_t)n0 * K, offB[0], K);
  }
  for (int t = 0; t < nk; ++t) {
    const int buf = t & 1;
    if (issuer) {
      if (t + 1 < nk) {
        tdm_load_tile(A  + (size_t)m0 * K + (t + 1) * 32, offA[buf ^ 1], K);
        tdm_load_tile(Bt + (size_t)n0 * K + (t + 1) * 32, offB[buf ^ 1], K);
        __builtin_amdgcn_s_wait_tensorcnt(2);  // pair for tile t is complete
      } else {
        __builtin_amdgcn_s_wait_tensorcnt(0);
      }
    }
    __syncthreads();   // publish tile t to all waves

    F16x16 af[4];
#pragma unroll
    for (int i = 0; i < 4; ++i) {
      int row = wm * 64 + i * 16 + l15;
      int c0  = half * 8;
      af[i].u[0] = *(const uint4*)&As[buf][row][c0];
      af[i].u[1] = *(const uint4*)&As[buf][row][16 + c0];
    }
    F16x16 bfr[2];
#pragma unroll
    for (int j = 0; j < 2; ++j) {
      int nrow = wn * 32 + j * 16 + l15;
      int c    = half * 8;
      bfr[j].u[0] = *(const uint4*)&Bs[buf][nrow][c];
      bfr[j].u[1] = *(const uint4*)&Bs[buf][nrow][c + 16];
    }
#pragma unroll
    for (int i = 0; i < 4; ++i)
#pragma unroll
      for (int j = 0; j < 2; ++j)
        acc[i][j] = wmma_bf16(af[i].v, bfr[j].v, acc[i][j]);
    __syncthreads();   // reads of buf done before DMA reuses it next iter
  }
#else
  // ---- fallback: manual VGPR staging ----
  const int ldRow = tid >> 1;          // 0..127
  const int ldSeg = (tid & 1) * 16;    // 0 or 16 (elements)
  for (int kt = 0; kt < K; kt += 32) {
    {
      const uint4* ga = (const uint4*)(A  + (size_t)(m0 + ldRow) * K + kt + ldSeg);
      const uint4* gb = (const uint4*)(Bt + (size_t)(n0 + ldRow) * K + kt + ldSeg);
      uint4 a0 = ga[0], a1 = ga[1];
      uint4 b0 = gb[0], b1 = gb[1];
      *(uint4*)&As[0][ldRow][ldSeg]     = a0;
      *(uint4*)&As[0][ldRow][ldSeg + 8] = a1;
      *(uint4*)&Bs[0][ldRow][ldSeg]     = b0;
      *(uint4*)&Bs[0][ldRow][ldSeg + 8] = b1;
    }
    __syncthreads();
    F16x16 af[4];
#pragma unroll
    for (int i = 0; i < 4; ++i) {
      int row = wm * 64 + i * 16 + l15;
      int c0  = half * 8;
      af[i].u[0] = *(const uint4*)&As[0][row][c0];
      af[i].u[1] = *(const uint4*)&As[0][row][16 + c0];
    }
    F16x16 bfr[2];
#pragma unroll
    for (int j = 0; j < 2; ++j) {
      int nrow = wn * 32 + j * 16 + l15;
      int c    = half * 8;
      bfr[j].u[0] = *(const uint4*)&Bs[0][nrow][c];
      bfr[j].u[1] = *(const uint4*)&Bs[0][nrow][c + 16];
    }
#pragma unroll
    for (int i = 0; i < 4; ++i)
#pragma unroll
      for (int j = 0; j < 2; ++j)
        acc[i][j] = wmma_bf16(af[i].v, bfr[j].v, acc[i][j]);
    __syncthreads();
  }
#endif

  // epilogue: C element (m,n): m = r + 8*half within tile, n = l15
#pragma unroll
  for (int i = 0; i < 4; ++i) {
#pragma unroll
    for (int j = 0; j < 2; ++j) {
      int n = n0 + wn * 32 + j * 16 + l15;
#pragma unroll
      for (int r = 0; r < 8; ++r) {
        int m = m0 + wm * 64 + i * 16 + r + 8 * half;
        float v = acc[i][j][r];
        if (MODE == 3) {
          ((float*)Out)[(size_t)m * N + n] = v;
        } else {
          int b = m >> 11, t = m & (T_ - 1);   // M rows = B*T tokens
          int h = n >> 6,  hd = n & (HD_ - 1); // N cols = H*HD
          __bf16* o = (__bf16*)Out;
          if (MODE == 0)
            o[(((size_t)(b * H_ + h)) * T_ + t) * HD_ + hd] = (__bf16)(v * 0.125f);
          else if (MODE == 1)
            o[(((size_t)(b * H_ + h)) * T_ + t) * HD_ + hd] = (__bf16)v;
          else // MODE == 2: V transposed per head
            o[(((size_t)(b * H_ + h)) * HD_ + hd) * T_ + t] = (__bf16)v;
        }
      }
    }
  }
}

// ---------------- flash attention (causal, online softmax) ----------------
// Q,K: bf16 [B,H,T,HD]; Vt: bf16 [B,H,HD,T]; Ctx out: bf16 [B,T,H,HD] == [B,T,D]
__global__ __launch_bounds__(256) void flash_attn_causal(
    const __bf16* __restrict__ Q, const __bf16* __restrict__ Kg,
    const __bf16* __restrict__ Vt, __bf16* __restrict__ Ctx)
{
  __shared__ __bf16 P[8][16][32]; // per-wave P staging (C-layout -> A-layout)

  const int tid  = threadIdx.x;
  const int lane = tid & 31, wave = tid >> 5;
  const int half = lane >> 4, l15 = lane & 15;
  const int bh = blockIdx.y;
  const int b = bh >> 4, h = bh & (H_ - 1);
  const int qb = blockIdx.x * 128 + wave * 16; // this wave's 16 query rows

  const __bf16* Qh = Q  + (size_t)bh * T_ * HD_;
  const __bf16* Kh = Kg + (size_t)bh * T_ * HD_;
  const __bf16* Vh = Vt + (size_t)bh * HD_ * T_;

  // Q fragments (16 rows x 64 hd) -> two A-fragments, resident in registers
  F16x16 qf[2];
#pragma unroll
  for (int kk = 0; kk < 2; ++kk) {
    const __bf16* p = Qh + (size_t)(qb + l15) * HD_ + kk * 32 + half * 8;
    qf[kk].u[0] = *(const uint4*)p;
    qf[kk].u[1] = *(const uint4*)(p + 16);
  }

  v8f o[4];
#pragma unroll
  for (int f = 0; f < 4; ++f) o[f] = zero8();
  float mrow[8], lsum[8];
#pragma unroll
  for (int r = 0; r < 8; ++r) { mrow[r] = -__builtin_inff(); lsum[r] = 0.f; }

  const int jmax = (qb + 15) >> 5; // last key tile (32 keys) touching the causal region
  for (int j = 0; j <= jmax; ++j) {
    const int k0 = j << 5;

    // prefetch next key/value tiles into cache while this tile computes
    if (j < jmax) {
      __builtin_prefetch(Kh + (size_t)(k0 + 32 + lane) * HD_, 0, 3);
      __builtin_prefetch(Vh + (size_t)(lane * 2) * T_ + k0 + 32, 0, 3);
    }

    // scores S = Q * K^T for 32 keys, split into two 16-key accumulators
    v8f s[2]; s[0] = zero8(); s[1] = zero8();
#pragma unroll
    for (int kk = 0; kk < 2; ++kk) {
#pragma unroll
      for (int g = 0; g < 2; ++g) {
        F16x16 kf;
        const __bf16* p = Kh + (size_t)(k0 + g * 16 + l15) * HD_ + kk * 32 + half * 8;
        kf.u[0] = *(const uint4*)p;
        kf.u[1] = *(const uint4*)(p + 16);
        s[g] = wmma_bf16(qf[kk].v, kf.v, s[g]);
      }
    }

    // causal mask + online softmax (row = r + 8*half, col = l15 within key group)
#pragma unroll
    for (int r = 0; r < 8; ++r) {
      const int qrow = qb + r + 8 * half;
      float s0 = (k0 + l15      <= qrow) ? s[0][r] : -__builtin_inff();
      float s1 = (k0 + 16 + l15 <= qrow) ? s[1][r] : -__builtin_inff();
      float t = fmaxf(s0, s1);
      t = fmaxf(t, __shfl_xor(t, 1, 32));
      t = fmaxf(t, __shfl_xor(t, 2, 32));
      t = fmaxf(t, __shfl_xor(t, 4, 32));
      t = fmaxf(t, __shfl_xor(t, 8, 32));
      const float mnew  = fmaxf(mrow[r], t);
      const float alpha = __expf(mrow[r] - mnew);
      const float p0 = __expf(s0 - mnew);
      const float p1 = __expf(s1 - mnew);
      float ps = p0 + p1;
      ps += __shfl_xor(ps, 1, 32);
      ps += __shfl_xor(ps, 2, 32);
      ps += __shfl_xor(ps, 4, 32);
      ps += __shfl_xor(ps, 8, 32);
      lsum[r] = lsum[r] * alpha + ps;
      mrow[r] = mnew;
#pragma unroll
      for (int f = 0; f < 4; ++f) o[f][r] *= alpha;
      P[wave][r + 8 * half][l15]      = (__bf16)p0;
      P[wave][r + 8 * half][16 + l15] = (__bf16)p1;
    }

    // repack P (16x32) as A-fragment from per-wave LDS (hw-ordered via DScnt)
    F16x16 pf;
    pf.u[0] = *(const uint4*)&P[wave][l15][half * 8];
    pf.u[1] = *(const uint4*)&P[wave][l15][16 + half * 8];

    // O += P * V  (K-dim = 32 keys, N = 64 hd in four 16-col fragments)
#pragma unroll
    for (int f = 0; f < 4; ++f) {
      F16x16 vf;
      const __bf16* p = Vh + (size_t)(f * 16 + l15) * T_ + k0 + half * 8;
      vf.u[0] = *(const uint4*)p;
      vf.u[1] = *(const uint4*)(p + 16);
      o[f] = wmma_bf16(pf.v, vf.v, o[f]);
    }
  }

  // normalize and write ctx as [B,T,H,HD] == [B,T,D]
#pragma unroll
  for (int r = 0; r < 8; ++r) {
    const int qrow = qb + r + 8 * half;
    const float inv = 1.0f / lsum[r];
#pragma unroll
    for (int f = 0; f < 4; ++f) {
      const int hd = f * 16 + l15;
      Ctx[(((size_t)b * T_ + qrow) * H_ + h) * HD_ + hd] = (__bf16)(o[f][r] * inv);
    }
  }
}

// ---------------- launch ----------------

extern "C" void kernel_launch(void* const* d_in, const int* in_sizes, int n_in,
                              void* d_out, int out_size, void* d_ws, size_t ws_size,
                              hipStream_t stream) {
  const float* z   = (const float*)d_in[0];
  const float* w_q = (const float*)d_in[1];
  const float* w_k = (const float*)d_in[2];
  const float* w_v = (const float*)d_in[3];
  const float* w_o = (const float*)d_in[4];

  const int M = B_ * T_;       // 4096
  const int ND = D_;           // 1024
  const size_t tokElems = (size_t)M * D_;   // 4M
  const size_t wElems   = (size_t)D_ * D_;  // 1M

  char* ws = (char*)d_ws;
  __bf16* zb  = (__bf16*)ws;                 ws += tokElems * 2;
  __bf16* wqb = (__bf16*)ws;                 ws += wElems * 2;
  __bf16* wkb = (__bf16*)ws;                 ws += wElems * 2;
  __bf16* wvb = (__bf16*)ws;                 ws += wElems * 2;
  __bf16* wob = (__bf16*)ws;                 ws += wElems * 2;
  __bf16* Qb  = (__bf16*)ws;                 ws += tokElems * 2;
  __bf16* Kb  = (__bf16*)ws;                 ws += tokElems * 2;
  __bf16* Vtb = (__bf16*)ws;                 ws += tokElems * 2;
  __bf16* ctx = (__bf16*)ws;                 ws += tokElems * 2;

  // convert activations and transposed bf16 weights
  cvt_f32_bf16<<<(int)(tokElems / 256), 256, 0, stream>>>(z, zb, (int)tokElems);
  cvt_w_transpose<<<(int)(wElems / 256), 256, 0, stream>>>(w_q, wqb);
  cvt_w_transpose<<<(int)(wElems / 256), 256, 0, stream>>>(w_k, wkb);
  cvt_w_transpose<<<(int)(wElems / 256), 256, 0, stream>>>(w_v, wvb);
  cvt_w_transpose<<<(int)(wElems / 256), 256, 0, stream>>>(w_o, wob);

  // Q/K/V projections
  dim3 gg(ND / 128, M / 128); // (8, 32)
  gemm_bf16_128x128<0><<<gg, 256, 0, stream>>>(zb, wqb, Qb,  M, ND, D_);
  gemm_bf16_128x128<1><<<gg, 256, 0, stream>>>(zb, wkb, Kb,  M, ND, D_);
  gemm_bf16_128x128<2><<<gg, 256, 0, stream>>>(zb, wvb, Vtb, M, ND, D_);

  // causal flash attention
  dim3 fg(T_ / 128, B_ * H_); // (16, 32)
  flash_attn_causal<<<fg, 256, 0, stream>>>(Qb, Kb, Vtb, ctx);

  // output projection -> fp32
  gemm_bf16_128x128<3><<<gg, 256, 0, stream>>>(ctx, wob, d_out, M, ND, D_);
}
